// Multi2GruMean_81449759801692
// MI455X (gfx1250) — matrix operand order repaired and checked by reading, not compiled, not run on hardware
//
#include <hip/hip_runtime.h>
#include <hip/hip_bf16.h>

typedef __bf16 bf16_t;
typedef __attribute__((ext_vector_type(16))) __bf16 v16bf;
typedef __attribute__((ext_vector_type(8)))  float  v8f;

// ---------------------------------------------------------------------------
// Fragment helpers (CDNA5 WMMA 16x16x32 bf16 layouts, wave32)
// A (16x32): lane -> M = lane%16, half = lane/16.
//   element e (0..15): K = 8*half + e + (e>=8 ? 8 : 0)
// B (32x16): lane%16 = N; element e holds K = 8*half + e + (e>=8 ? 8 : 0), as A
// C/D (16x16 f32): lane%16 = N; VGPR r holds M = r + 8*(lane/16)
// ---------------------------------------------------------------------------
__device__ __forceinline__ int amap_k(int e, int half) {
  return 8 * half + e + ((e >= 8) ? 8 : 0);
}

__device__ __forceinline__ v16bf load_a_bf16(const bf16_t* s, int m, int stride,
                                             int k0, int half) {
  v16bf a;
#pragma unroll
  for (int e = 0; e < 16; ++e) a[e] = s[m * stride + k0 + amap_k(e, half)];
  return a;
}

__device__ __forceinline__ v8f wmma_bf16(v16bf a, v16bf b, v8f c) {
  return __builtin_amdgcn_wmma_f32_16x16x32_bf16(false, a, false, b, (short)0,
                                                 c, false, false);
}

// ---------------------------------------------------------------------------
// Pack W (3H x Ksrc fp32, row-major) as B fragments: [24 nt][KT kt][32 ln][16 e]
// dst element = W.T[k][n] = W[n][k], zero-padded for k >= Ksrc.
// ---------------------------------------------------------------------------
__global__ void pack_b_kernel(const float* __restrict__ W, bf16_t* __restrict__ dst,
                              int KT, int Ksrc) {
  int total = 24 * KT * 512;
  for (int idx = blockIdx.x * blockDim.x + threadIdx.x; idx < total;
       idx += gridDim.x * blockDim.x) {
    int e    = idx & 15;
    int lane = (idx >> 4) & 31;
    int fi   = idx >> 9;
    int kt   = fi % KT;
    int nt   = fi / KT;
    int n = nt * 16 + (lane & 15);
    int k = kt * 32 + 8 * (lane >> 4) + e + ((e >= 8) ? 8 : 0);
    float v = (k < Ksrc) ? W[(long)n * Ksrc + k] : 0.f;
    dst[idx] = (bf16_t)v;
  }
}

// ---------------------------------------------------------------------------
// Persistent fused BiGRU kernel: one block = 16 sequences, one direction,
// full time loop. blockIdx.x = sequence block, blockIdx.y = direction.
// MODE 0: x_t = embedding gather (word level). MODE 1: x_t = xf + xb rows.
// Compile-time KTX/EPAD keep the hoisted A-fragment arrays in VGPRs.
// ---------------------------------------------------------------------------
template <int MODE, int EPAD, int KTX>
__global__ __launch_bounds__(256) void gru_wmma_kernel(
    int T, int Esrc,
    const int* __restrict__ tokens, const float* __restrict__ embed,
    const float* __restrict__ xin_f, const float* __restrict__ xin_b,
    const int* __restrict__ lens,
    const bf16_t* __restrict__ pk_wih, const bf16_t* __restrict__ pk_whh,
    const float* __restrict__ bih0, const float* __restrict__ bhh0,
    const float* __restrict__ bih1, const float* __restrict__ bhh1,
    float* __restrict__ out0, float* __restrict__ out1) {
  __shared__ float  s_g [16 * 256];   // 16 KB  (r,z: xi+hg summed)
  __shared__ float  s_xn[16 * 128];   // 8 KB   (n: xi)
  __shared__ float  s_hn[16 * 128];   // 8 KB   (n: hg)
  __shared__ float  s_h [16 * 128];   // 8 KB   fp32 master h
  __shared__ bf16_t s_hb[16 * 128];   // 4 KB   bf16 shadow h
  __shared__ bf16_t s_x [16 * 224];   // 7 KB
  __shared__ int    s_len[16];

  const int tid  = threadIdx.x;
  const int lane = tid & 31;
  const int wave = tid >> 5;
  const int half = lane >> 4;
  const int m16  = lane & 15;
  const int dir  = blockIdx.y;
  const int seq0 = blockIdx.x * 16;

  const float* bih = dir ? bih1 : bih0;
  const float* bhh = dir ? bhh1 : bhh0;
  float*       out = dir ? out1 : out0;

  for (int i = tid; i < 16 * 128; i += 256) {
    s_h[i]  = 0.f;
    s_hb[i] = (bf16_t)0.f;
  }
  if (tid < 16) s_len[tid] = lens[seq0 + tid];

  // Pin Whh B fragments in VGPRs for the whole recurrence (3 nt x 4 kt x 8 VGPR)
  v16bf whh[3][4];
#pragma unroll
  for (int i = 0; i < 3; ++i) {
    int nt = wave * 3 + i;
#pragma unroll
    for (int kt = 0; kt < 4; ++kt) {
      long frag = (long)(dir * 24 + nt) * 4 + kt;
      whh[i][kt] = *(const v16bf*)(pk_whh + frag * 512 + lane * 16);
    }
  }
  __syncthreads();

  for (int step = 0; step < T; ++step) {
    const int t = dir ? (T - 1 - step) : step;  // reverse == reversed walk

    // ---- stage x_t into LDS as bf16 -----------------------------------
    {
      int m = tid >> 4, kk = tid & 15;
      if (MODE == 0) {
        int tok = tokens[(long)(seq0 + m) * T + t];
        const float* erow = embed + (long)tok * Esrc;
#pragma unroll
        for (int k = kk; k < EPAD; k += 16)
          s_x[m * EPAD + k] = (k < Esrc) ? (bf16_t)erow[k] : (bf16_t)0.f;
      } else {
        long row = ((long)(seq0 + m) * T + t) * 128;
#pragma unroll
        for (int k = kk; k < EPAD; k += 16)
          s_x[m * EPAD + k] = (bf16_t)(xin_f[row + k] + xin_b[row + k]);
      }
    }
    __syncthreads();

    // ---- hoist A fragments (shared across this wave's 3 N-tiles) ------
    v16bf ax[KTX];
#pragma unroll
    for (int kt = 0; kt < KTX; ++kt)
      ax[kt] = load_a_bf16(s_x, m16, EPAD, kt * 32, half);
    v16bf ah[4];
#pragma unroll
    for (int kt = 0; kt < 4; ++kt)
      ah[kt] = load_a_bf16(s_hb, m16, 128, kt * 32, half);

    // ---- WMMA phase: xi = x*Wih^T, hg = h*Whh^T -----------------------
#pragma unroll
    for (int i = 0; i < 3; ++i) {
      int nt = wave * 3 + i;
      v8f accx = {0.f, 0.f, 0.f, 0.f, 0.f, 0.f, 0.f, 0.f};
#pragma unroll
      for (int kt = 0; kt < KTX; ++kt) {
        long frag = (long)(dir * 24 + nt) * KTX + kt;
        v16bf b = *(const v16bf*)(pk_wih + frag * 512 + lane * 16);
        accx = wmma_bf16(ax[kt], b, accx);
      }
      v8f acch = {0.f, 0.f, 0.f, 0.f, 0.f, 0.f, 0.f, 0.f};
#pragma unroll
      for (int kt = 0; kt < 4; ++kt)
        acch = wmma_bf16(ah[kt], whh[i][kt], acch);

      if (nt < 16) {            // wave-uniform branch (EXEC stays full)
        int n = nt * 16 + m16;  // column in [0,256): r gate then z gate
#pragma unroll
        for (int r = 0; r < 8; ++r)
          s_g[(r + 8 * half) * 256 + n] = accx[r] + acch[r];
      } else {
        int n = (nt - 16) * 16 + m16;  // column in [0,128): n gate
#pragma unroll
        for (int r = 0; r < 8; ++r) {
          s_xn[(r + 8 * half) * 128 + n] = accx[r];
          s_hn[(r + 8 * half) * 128 + n] = acch[r];
        }
      }
    }
    __syncthreads();

    // ---- pointwise GRU gates (fp32), update fp32 h + bf16 shadow ------
    for (int idx = tid; idx < 16 * 128; idx += 256) {
      int m = idx >> 7, j = idx & 127;
      float gr = s_g[m * 256 + j]       + bih[j]       + bhh[j];
      float gz = s_g[m * 256 + 128 + j] + bih[128 + j] + bhh[128 + j];
      float xn = s_xn[idx] + bih[256 + j];
      float hn = s_hn[idx] + bhh[256 + j];
      float rg = 1.f / (1.f + __expf(-gr));
      float zg = 1.f / (1.f + __expf(-gz));
      float ng = tanhf(xn + rg * hn);
      float h  = s_h[idx];
      float hnew = (1.f - zg) * ng + zg * h;
      float hsel = (t < s_len[m]) ? hnew : h;
      s_h[idx]  = hsel;
      s_hb[idx] = (bf16_t)hsel;
    }
    __syncthreads();
  }

  // write 0.5*h_last (so repf + repb == bigru average)
  for (int i = tid; i < 16 * 128; i += 256) {
    int m = i >> 7, j = i & 127;
    out[(long)(seq0 + m) * 128 + j] = 0.5f * s_h[i];
  }
}

// ---------------------------------------------------------------------------
// Head: FC1 + SELU + FC2, then p_stars = mean over R. out = [16 p][160 r].
// ---------------------------------------------------------------------------
__global__ __launch_bounds__(256) void head_kernel(
    const float* __restrict__ revf, const float* __restrict__ revb,
    const float* __restrict__ fc1w, const float* __restrict__ fc1b,
    const float* __restrict__ fc2w, const float* __restrict__ fc2b,
    float* __restrict__ out) {
  __shared__ float s_r[160];
  const int tid = threadIdx.x;
  if (tid < 160) {
    const float kScale = 1.0507009873554805f, kAlpha = 1.6732632423543772f;
    float rstar = fc2b[0];
    for (int j = 0; j < 64; ++j) {
      float acc = fc1b[j];
      for (int k = 0; k < 128; ++k)
        acc += (revf[tid * 128 + k] + revb[tid * 128 + k]) * fc1w[j * 128 + k];
      float h = (acc > 0.f) ? kScale * acc : kScale * kAlpha * (__expf(acc) - 1.f);
      rstar += h * fc2w[j];
    }
    s_r[tid] = rstar;
    out[16 + tid] = rstar;
  }
  __syncthreads();
  if (tid < 16) {
    float s = 0.f;
    for (int i = 0; i < 10; ++i) s += s_r[tid * 10 + i];
    out[tid] = s * 0.1f;
  }
}

// ---------------------------------------------------------------------------
extern "C" void kernel_launch(void* const* d_in, const int* in_sizes, int n_in,
                              void* d_out, int out_size, void* d_ws, size_t ws_size,
                              hipStream_t stream) {
  (void)in_sizes; (void)n_in; (void)out_size; (void)ws_size;
  const int*   tokens   = (const int*)d_in[0];
  const int*   slen     = (const int*)d_in[1];
  const int*   scnt     = (const int*)d_in[2];
  const float* embed    = (const float*)d_in[3];
  const float* wWih_f   = (const float*)d_in[4];
  const float* wWhh_f   = (const float*)d_in[5];
  const float* wbih_f   = (const float*)d_in[6];
  const float* wbhh_f   = (const float*)d_in[7];
  const float* wWih_b   = (const float*)d_in[8];
  const float* wWhh_b   = (const float*)d_in[9];
  const float* wbih_b   = (const float*)d_in[10];
  const float* wbhh_b   = (const float*)d_in[11];
  const float* sWih_f   = (const float*)d_in[12];
  const float* sWhh_f   = (const float*)d_in[13];
  const float* sbih_f   = (const float*)d_in[14];
  const float* sbhh_f   = (const float*)d_in[15];
  const float* sWih_b   = (const float*)d_in[16];
  const float* sWhh_b   = (const float*)d_in[17];
  const float* sbih_b   = (const float*)d_in[18];
  const float* sbhh_b   = (const float*)d_in[19];
  const float* fc1w     = (const float*)d_in[20];
  const float* fc1b     = (const float*)d_in[21];
  const float* fc2w     = (const float*)d_in[22];
  const float* fc2b     = (const float*)d_in[23];
  float* out = (float*)d_out;

  // ---- workspace carve-out (256B aligned) ----
  char* base = (char*)d_ws;
  size_t off = 0;
  auto carve = [&](size_t bytes) -> char* {
    off = (off + 255) & ~(size_t)255;
    char* p = base + off;
    off += bytes;
    return p;
  };
  const size_t FR_W = (size_t)24 * 7 * 512;  // frags per dir, word Wih
  const size_t FR_4 = (size_t)24 * 4 * 512;  // frags per dir, K=128 matrices
  bf16_t* pk_wih_w = (bf16_t*)carve(2 * FR_W * sizeof(bf16_t));
  bf16_t* pk_whh_w = (bf16_t*)carve(2 * FR_4 * sizeof(bf16_t));
  bf16_t* pk_wih_s = (bf16_t*)carve(2 * FR_4 * sizeof(bf16_t));
  bf16_t* pk_whh_s = (bf16_t*)carve(2 * FR_4 * sizeof(bf16_t));
  float* repf = (float*)carve((size_t)3200 * 128 * sizeof(float));
  float* repb = (float*)carve((size_t)3200 * 128 * sizeof(float));
  float* revf = (float*)carve((size_t)160 * 128 * sizeof(float));
  float* revb = (float*)carve((size_t)160 * 128 * sizeof(float));

  // ---- pack weights to bf16 B fragments ----
  pack_b_kernel<<<336, 256, 0, stream>>>(wWih_f, pk_wih_w,        7, 200);
  pack_b_kernel<<<336, 256, 0, stream>>>(wWih_b, pk_wih_w + FR_W, 7, 200);
  pack_b_kernel<<<192, 256, 0, stream>>>(wWhh_f, pk_whh_w,        4, 128);
  pack_b_kernel<<<192, 256, 0, stream>>>(wWhh_b, pk_whh_w + FR_4, 4, 128);
  pack_b_kernel<<<192, 256, 0, stream>>>(sWih_f, pk_wih_s,        4, 128);
  pack_b_kernel<<<192, 256, 0, stream>>>(sWih_b, pk_wih_s + FR_4, 4, 128);
  pack_b_kernel<<<192, 256, 0, stream>>>(sWhh_f, pk_whh_s,        4, 128);
  pack_b_kernel<<<192, 256, 0, stream>>>(sWhh_b, pk_whh_s + FR_4, 4, 128);

  // ---- word-level BiGRU: 3200 seqs, T=64, E=200 (pad 224) ----
  gru_wmma_kernel<0, 224, 7><<<dim3(200, 2), 256, 0, stream>>>(
      /*T=*/64, /*Esrc=*/200,
      tokens, embed, nullptr, nullptr, slen,
      pk_wih_w, pk_whh_w, wbih_f, wbhh_f, wbih_b, wbhh_b, repf, repb);

  // ---- sentence-level BiGRU: 160 seqs, T=20, E=128 ----
  gru_wmma_kernel<1, 128, 4><<<dim3(10, 2), 256, 0, stream>>>(
      /*T=*/20, /*Esrc=*/128,
      nullptr, nullptr, repf, repb, scnt,
      pk_wih_s, pk_whh_s, sbih_f, sbhh_f, sbih_b, sbhh_b, revf, revb);

  // ---- head ----
  head_kernel<<<1, 256, 0, stream>>>(revf, revb, fc1w, fc1b, fc2w, fc2b, out);
}
